// ConvPDEncoder_10849087390295
// MI455X (gfx1250) — matrix operand order repaired and checked by reading, not compiled, not run on hardware
//
#include <hip/hip_runtime.h>

// Problem constants (match reference)
constexpr int Bc   = 2;
constexpr int Nc   = 4096;
constexpr int Gc   = 2048;
constexpr int COUT = 8;

constexpr int BDIM  = 256;  // 8 waves of 32
constexpr int TILE  = 128;  // macro-tile edge (g and h)
constexpr int KSTEP = 32;   // bf16 WMMA K
constexpr int LDK   = 40;   // padded LDS row length (elems): 64B data + 16B pad

typedef __attribute__((ext_vector_type(16))) __bf16 v16bf;
typedef __attribute__((ext_vector_type(8)))  float  v8f;

union Frag {
  v16bf v;
  uint4 q[2];
};

__device__ __forceinline__ unsigned short f2bf(float f) {
  unsigned int u = __float_as_uint(f);
  u += 0x7fffu + ((u >> 16) & 1u);   // round-to-nearest-even
  return (unsigned short)(u >> 16);
}

// ---------------------------------------------------------------------------
// Phase 1: materialize Wt[b][g][n] = bf16(w), WZt[b][g][n] = bf16(w*z)
// (transposed layout: row g, contiguous n -> NT-GEMM feeds are coalesced)
// Also copies x_grid into the head of d_out (tuple output element 0).
// ---------------------------------------------------------------------------
__global__ void __launch_bounds__(BDIM) rbf_phase1(
    const float* __restrict__ xz, const float* __restrict__ z,
    const float* __restrict__ xg, const float* __restrict__ ls,
    unsigned short* __restrict__ Wt, unsigned short* __restrict__ WZt,
    float* __restrict__ grid_out) {
  const int bid = blockIdx.x;            // b*Gc + g
  const int b   = bid / Gc;
  const int g   = bid - b * Gc;
  const int tid = threadIdx.x;

  const int gcopy = bid * BDIM + tid;    // first 8 blocks cover G=2048
  if (gcopy < Gc) grid_out[gcopy] = xg[gcopy];

  const float coef = -0.5f / __expf(2.0f * ls[0]);   // -0.5 / exp(2*log_scale)
  const float gx   = xg[g];
  const long  rowbase = (long)bid * Nc;

  for (int n = tid; n < Nc; n += BDIM) {
    const float dx = xz[b * Nc + n] - gx;
    const float w  = __expf(coef * dx * dx);
    Wt [rowbase + n] = f2bf(w);
    WZt[rowbase + n] = f2bf(w * z[b * Nc + n]);
  }
}

// ---------------------------------------------------------------------------
// Phase 2: per batch, C0 = Wt*Wt^T, C1 = WZt*Wt^T (both G x G, SYMMETRIC).
// Only lower-triangular macro-tiles (g0 >= h0) are computed; each tile is
// written to (g,h) directly and to (h,g) via an LDS-transposed, coalesced
// store path. Fused epilogue:
// out[o,g,h] = W[0,o]*(g==h) + W[1,o]*C0 + W[2,o]*(C1/(C0+1e-8)) + b[o]
// ---------------------------------------------------------------------------
__global__ void __launch_bounds__(BDIM) gemm_phase2(
    const unsigned short* __restrict__ Wt, const unsigned short* __restrict__ WZt,
    const float* __restrict__ Wlin, const float* __restrict__ bias,
    float* __restrict__ out) {
  __shared__ __align__(16) unsigned short ldsA0[TILE * LDK];
  __shared__ __align__(16) unsigned short ldsA1[TILE * LDK];
  __shared__ __align__(16) unsigned short ldsB [TILE * LDK];
  __shared__ float ldsC[8 * 256];   // per-wave 16x16 staging: density
  __shared__ float ldsR[8 * 256];   // per-wave 16x16 staging: ratio

  // Triangular block decode: row block `by`, col block `bx`, bx <= by.
  int idx = blockIdx.x;
  int by = 0;
  while (idx >= by + 1) { idx -= (by + 1); ++by; }
  const int bx = idx;

  const int g0 = by * TILE;   // row range (g)
  const int h0 = bx * TILE;   // col range (h), h0 <= g0
  const int b  = blockIdx.y;

  const int tid  = threadIdx.x;
  const int wid  = tid >> 5;
  const int lane = tid & 31;
  const int lh   = lane >> 4;   // lane half (0/1)
  const int ln   = lane & 15;

  const int wm = wid & 1;       // 2 row groups of 64 (g)
  const int wn = wid >> 1;      // 4 col groups of 32 (h)

  const unsigned short* WtB  = Wt  + (long)b * Gc * Nc;
  const unsigned short* WZtB = WZt + (long)b * Gc * Nc;

  v8f acc0[8], acc1[8];
  #pragma unroll
  for (int t = 0; t < 8; ++t) {
    #pragma unroll
    for (int i = 0; i < 8; ++i) { acc0[t][i] = 0.0f; acc1[t][i] = 0.0f; }
  }

  const int frow = tid >> 1;        // LDS fill: row 0..127
  const int fk   = (tid & 1) * 16;  // k offset (elems): 0 or 16

  for (int k = 0; k < Nc; k += KSTEP) {
    __syncthreads();
    {
      const uint4* sA0 = (const uint4*)(WtB  + (long)(g0 + frow) * Nc + k + fk);
      const uint4* sA1 = (const uint4*)(WZtB + (long)(g0 + frow) * Nc + k + fk);
      const uint4* sB  = (const uint4*)(WtB  + (long)(h0 + frow) * Nc + k + fk);
      uint4 a0lo = sA0[0], a0hi = sA0[1];
      uint4 a1lo = sA1[0], a1hi = sA1[1];
      uint4 blo  = sB[0],  bhi  = sB[1];
      *(uint4*)&ldsA0[frow * LDK + fk]     = a0lo;
      *(uint4*)&ldsA0[frow * LDK + fk + 8] = a0hi;
      *(uint4*)&ldsA1[frow * LDK + fk]     = a1lo;
      *(uint4*)&ldsA1[frow * LDK + fk + 8] = a1hi;
      *(uint4*)&ldsB [frow * LDK + fk]     = blo;
      *(uint4*)&ldsB [frow * LDK + fk + 8] = bhi;
    }
    __syncthreads();

    // B fragments: 32x16 (KxN). lanes 0-15: K=0..15 @ N=lane; lanes 16-31: K=16..31
    Frag bf[2];
    #pragma unroll
    for (int tn = 0; tn < 2; ++tn) {
      const int col = wn * 32 + tn * 16 + ln;
      const int kb  = lh * 16;
      bf[tn].q[0] = *(const uint4*)&ldsB[col * LDK + kb];
      bf[tn].q[1] = *(const uint4*)&ldsB[col * LDK + kb + 8];
    }

    #pragma unroll
    for (int tm = 0; tm < 4; ++tm) {
      // A fragments: 16x32 (MxK). lanes 0-15: K 0..15; lanes 16-31: K 16..31
      const int row = wm * 64 + tm * 16 + ln;
      const int kb0 = lh ? 16 : 0;
      const int kb1 = lh ? 24 : 8;
      Frag a0, a1;
      a0.q[0] = *(const uint4*)&ldsA0[row * LDK + kb0];
      a0.q[1] = *(const uint4*)&ldsA0[row * LDK + kb1];
      a1.q[0] = *(const uint4*)&ldsA1[row * LDK + kb0];
      a1.q[1] = *(const uint4*)&ldsA1[row * LDK + kb1];
      #pragma unroll
      for (int tn = 0; tn < 2; ++tn) {
        acc0[tm * 2 + tn] = __builtin_amdgcn_wmma_f32_16x16x32_bf16(
            false, a0.v, false, bf[tn].v, (short)0, acc0[tm * 2 + tn], false, false);
        acc1[tm * 2 + tn] = __builtin_amdgcn_wmma_f32_16x16x32_bf16(
            false, a1.v, false, bf[tn].v, (short)0, acc1[tm * 2 + tn], false, false);
      }
    }
  }

  // Fused epilogue: normalization + identity channel + 1x1 linear + bias.
  float w0[COUT], w1[COUT], w2[COUT], bo[COUT];
  #pragma unroll
  for (int o = 0; o < COUT; ++o) {
    w0[o] = Wlin[0 * COUT + o];
    w1[o] = Wlin[1 * COUT + o];
    w2[o] = Wlin[2 * COUT + o];
    bo[o] = bias[o];
  }

  float* ldsCw = &ldsC[wid * 256];  // wave-private 16x16 staging
  float* ldsRw = &ldsR[wid * 256];

  #pragma unroll
  for (int tm = 0; tm < 4; ++tm) {
    #pragma unroll
    for (int tn = 0; tn < 2; ++tn) {
      const int t   = tm * 2 + tn;
      const int gT0 = g0 + wm * 64 + tm * 16;   // tile row base (g)
      const int hT0 = h0 + wn * 32 + tn * 16;   // tile col base (h)

      // --- direct (g,h) writes, h contiguous across lanes ---
      #pragma unroll
      for (int i = 0; i < 8; ++i) {
        const int   m  = i + 8 * lh;            // C layout: VGPR i -> M = i + 8*laneHalf
        const int   g  = gT0 + m;
        const int   h  = hT0 + ln;              // N = lane & 15
        const float c0 = acc0[t][i];
        const float r  = acc1[t][i] / (c0 + 1e-8f);
        const float id = (g == h) ? 1.0f : 0.0f;
        // stage for transposed path (wave-private slab; LDS ops are in-order)
        ldsCw[m * 16 + ln] = c0;
        ldsRw[m * 16 + ln] = r;
        #pragma unroll
        for (int o = 0; o < COUT; ++o) {
          out[((long)(b * COUT + o) * Gc + g) * Gc + h] =
              w0[o] * id + w1[o] * c0 + w2[o] * r + bo[o];
        }
      }

      // --- mirrored (h,g) writes: read tile transposed so g is contiguous ---
      #pragma unroll
      for (int i = 0; i < 8; ++i) {
        const int   m2 = i + 8 * lh;            // local h (output row)
        const int   rowg = hT0 + m2;            // output row index (h side)
        const int   colg = gT0 + ln;            // output col index (g side), contiguous
        const float c0 = ldsCw[ln * 16 + m2];   // C[g=gT0+ln][h=hT0+m2] == C[rowg][colg]
        const float r  = ldsRw[ln * 16 + m2];
        const float id = (rowg == colg) ? 1.0f : 0.0f;
        #pragma unroll
        for (int o = 0; o < COUT; ++o) {
          out[((long)(b * COUT + o) * Gc + rowg) * Gc + colg] =
              w0[o] * id + w1[o] * c0 + w2[o] * r + bo[o];
        }
      }
    }
  }
}

extern "C" void kernel_launch(void* const* d_in, const int* in_sizes, int n_in,
                              void* d_out, int out_size, void* d_ws, size_t ws_size,
                              hipStream_t stream) {
  const float* xz = (const float*)d_in[0];   // (B,N,1)
  const float* z  = (const float*)d_in[1];   // (B,N,1)
  const float* xg = (const float*)d_in[2];   // (G,)
  const float* ls = (const float*)d_in[3];   // scalar log_scale
  const float* Wl = (const float*)d_in[4];   // (3, COUT)
  const float* bi = (const float*)d_in[5];   // (COUT,)

  float* outp = (float*)d_out;               // [x_grid (G) | out (B,COUT,G,G)]

  unsigned short* Wt  = (unsigned short*)d_ws;                 // bf16, B*G*N
  unsigned short* WZt = Wt + (size_t)Bc * Gc * Nc;             // bf16, B*G*N

  rbf_phase1<<<dim3(Bc * Gc), BDIM, 0, stream>>>(xz, z, xg, ls, Wt, WZt, outp);

  constexpr int NBLK = (Gc / TILE);                 // 16
  constexpr int NTRI = NBLK * (NBLK + 1) / 2;       // 136 lower-triangular tiles
  dim3 grid2(NTRI, Bc);
  gemm_phase2<<<grid2, BDIM, 0, stream>>>(Wt, WZt, Wl, bi, outp + Gc);
}
